// InterClassConstraint_15169824489717
// MI455X (gfx1250) — matrix operand order repaired and checked by reading, not compiled, not run on hardware
//
#include <hip/hip_runtime.h>

// ---------- problem constants ----------
#define NCLS 4096
#define FEAT 2048

// ---------- plain POD vector types (avoid HIP vector-type ctors) ----------
typedef __attribute__((ext_vector_type(4)))  unsigned int u32x4;
typedef __attribute__((ext_vector_type(4)))  float        f32x4;
typedef __attribute__((ext_vector_type(16))) __bf16       v16bf;
typedef __attribute__((ext_vector_type(8)))  float        v8f;

union Frag16 {            // 32 bytes: one WMMA bf16 A/B fragment per lane
    v16bf v;
    u32x4 q[2];
};

// round-to-nearest-even f32 -> bf16 (bit pattern in ushort)
static __device__ __forceinline__ unsigned short f2bf(float x) {
    unsigned int u = __float_as_uint(x);
    u += 0x7FFFu + ((u >> 16) & 1u);
    return (unsigned short)(u >> 16);
}

// =====================================================================
// Kernel 1: per-row prep.  One block (256 thr) per row i:
//   - block-reduce ||p_i||^2 and <p_i, w_i> in f32
//   - store bf16(p_i * inv_norm) and bf16(w_i) into workspace
//   - term1[i] = (1 - b[i]) * <p_hat_i, w_i>   (exact f32 path)
// =====================================================================
__global__ __launch_bounds__(256)
void icc_prep(const float* __restrict__ B,
              const float* __restrict__ P,
              const float* __restrict__ W,
              unsigned short* __restrict__ pB,
              unsigned short* __restrict__ wB,
              float* __restrict__ term1) {
    const int i = blockIdx.x;
    const int t = threadIdx.x;
    const float* prow = P + (size_t)i * FEAT;
    const float* wrow = W + (size_t)i * FEAT;

    f32x4 pv[2], wv[2];
    float ss = 0.f, dt = 0.f;
    #pragma unroll
    for (int q = 0; q < 2; ++q) {
        int idx = t * 2 + q;                       // float4 index, 512 per row
        pv[q] = ((const f32x4*)prow)[idx];
        wv[q] = ((const f32x4*)wrow)[idx];
        #pragma unroll
        for (int e = 0; e < 4; ++e) {
            ss += pv[q][e] * pv[q][e];
            dt += pv[q][e] * wv[q][e];
        }
    }

    __shared__ float s_ss[256];
    __shared__ float s_dt[256];
    s_ss[t] = ss; s_dt[t] = dt;
    __syncthreads();
    #pragma unroll
    for (int s = 128; s > 0; s >>= 1) {
        if (t < s) { s_ss[t] += s_ss[t + s]; s_dt[t] += s_dt[t + s]; }
        __syncthreads();
    }
    const float inv = 1.0f / fmaxf(sqrtf(s_ss[0]), 1e-12f);

    #pragma unroll
    for (int q = 0; q < 2; ++q) {
        int base = (t * 2 + q) * 4;                // element index
        unsigned short pb[4], wb[4];
        #pragma unroll
        for (int e = 0; e < 4; ++e) {
            pb[e] = f2bf(pv[q][e] * inv);
            wb[e] = f2bf(wv[q][e]);
        }
        unsigned int p01 = (unsigned int)pb[0] | ((unsigned int)pb[1] << 16);
        unsigned int p23 = (unsigned int)pb[2] | ((unsigned int)pb[3] << 16);
        unsigned int w01 = (unsigned int)wb[0] | ((unsigned int)wb[1] << 16);
        unsigned int w23 = (unsigned int)wb[2] | ((unsigned int)wb[3] << 16);
        ((unsigned long long*)(pB + (size_t)i * FEAT))[base >> 2] =
            (unsigned long long)p01 | ((unsigned long long)p23 << 32);
        ((unsigned long long*)(wB + (size_t)i * FEAT))[base >> 2] =
            (unsigned long long)w01 | ((unsigned long long)w23 << 32);
    }

    if (t == 0) {
        float diag = s_dt[0] * inv;
        term1[i] = (1.0f - B[i]) * diag;
    }
}

// =====================================================================
// Kernel 2: tiled bf16 WMMA GEMM, async-to-LDS double buffered,
// fused epilogue reduction.
// Block = 256 threads = 8 waves; tile 128(i) x 128(j); BK = 64.
// Wave w: rows (w&3)*32, cols (w>>2)*64  ->  2x4 grid of 16x16 tiles.
// =====================================================================
#define BM 128
#define BK 64
#define LDP 72          // LDS row stride in ushorts (pad 8 -> 144B, 16B aligned)

__global__ __launch_bounds__(256)
void icc_gemm(const unsigned short* __restrict__ pB,
              const unsigned short* __restrict__ wB,
              const float* __restrict__ term1,
              const float* __restrict__ B,
              float* __restrict__ partials) {
    __shared__ unsigned short Asm[2][BM][LDP];
    __shared__ unsigned short Bsm[2][BM][LDP];
    __shared__ float t1s[BM];
    __shared__ float bm1s[BM];
    __shared__ float red[256];

    const int t    = threadIdx.x;
    const int lane = t & 31;
    const int wv   = t >> 5;
    const int l15  = lane & 15;
    const int half = lane >> 4;
    const int mr0  = (wv & 3) * 32;    // wave row offset inside tile
    const int nc0  = (wv >> 2) * 64;   // wave col offset inside tile

    const int i0 = blockIdx.y * BM;
    const int j0 = blockIdx.x * BM;

    if (t < BM) {
        t1s[t]  = term1[i0 + t];
        bm1s[t] = B[j0 + t] - 1.0f;
    }

    // per-thread element offsets for the 4 chunks this thread copies
    // chunk q covers tile row r = (t+q*256)>>3, element col c = ((t+q*256)&7)*8
    int voffA[4], voffB[4], rr[4], cc[4];
    #pragma unroll
    for (int q = 0; q < 4; ++q) {
        int lin = t + q * 256;
        rr[q] = lin >> 3;
        cc[q] = (lin & 7) * 8;
        voffA[q] = (i0 + rr[q]) * FEAT + cc[q];
        voffB[q] = (j0 + rr[q]) * FEAT + cc[q];
    }

    // issue one K-tile's worth of async global->LDS copies into buffer `buf`
    auto issue = [&](int buf, int k0) {
        #pragma unroll
        for (int q = 0; q < 4; ++q) {
            const unsigned short* ag = pB + (size_t)(voffA[q] + k0);
            const unsigned short* bg = wB + (size_t)(voffB[q] + k0);
            // LDS byte offset = low 32 bits of flat shared-space address
            unsigned la = (unsigned)(unsigned long long)&Asm[buf][rr[q]][cc[q]];
            unsigned lb = (unsigned)(unsigned long long)&Bsm[buf][rr[q]][cc[q]];
            asm volatile("global_load_async_to_lds_b128 %0, %1, off"
                         :: "v"(la), "v"(ag) : "memory");
            asm volatile("global_load_async_to_lds_b128 %0, %1, off"
                         :: "v"(lb), "v"(bg) : "memory");
        }
    };

    v8f acc[2][4] = {};

    issue(0, 0);
    asm volatile("s_wait_asynccnt 0" ::: "memory");
    __syncthreads();

    int buf = 0;
    for (int k0 = 0; k0 < FEAT; k0 += BK, buf ^= 1) {
        if (k0 + BK < FEAT) issue(buf ^ 1, k0 + BK);   // overlap next tile

        // ---- two K=32 WMMA steps on current LDS buffer ----
        #pragma unroll
        for (int kk = 0; kk < BK; kk += 32) {
            Frag16 af[2];
            #pragma unroll
            for (int rt = 0; rt < 2; ++rt) {
                // A 16x32 bf16 layout: lane m=l15, K = half*8+e (e<8), 16+half*8+e
                const unsigned short* base =
                    &Asm[buf][mr0 + rt * 16 + l15][kk + half * 8];
                af[rt].q[0] = *(const u32x4*)base;
                af[rt].q[1] = *(const u32x4*)(base + 16);
            }
            Frag16 bfar[4];
            #pragma unroll
            for (int ct = 0; ct < 4; ++ct) {
                // B 32x16 bf16 layout: lane n=l15, K = half*8+e (e<8), 16+half*8+e
                const unsigned short* base =
                    &Bsm[buf][nc0 + ct * 16 + l15][kk + half * 8];
                bfar[ct].q[0] = *(const u32x4*)base;
                bfar[ct].q[1] = *(const u32x4*)(base + 16);
            }
            #pragma unroll
            for (int rt = 0; rt < 2; ++rt)
                #pragma unroll
                for (int ct = 0; ct < 4; ++ct)
                    acc[rt][ct] = __builtin_amdgcn_wmma_f32_16x16x32_bf16(
                        false, af[rt].v, false, bfar[ct].v,
                        (short)0, acc[rt][ct], false, false);
        }

        asm volatile("s_wait_asynccnt 0" ::: "memory");  // next tile landed
        __syncthreads();                                 // all waves done w/ cur
    }

    // ---- fused epilogue: relu(term1[i] + s*(b[j]-1)), mask diag, reduce ----
    float lsum = 0.f;
    #pragma unroll
    for (int rt = 0; rt < 2; ++rt) {
        #pragma unroll
        for (int ct = 0; ct < 4; ++ct) {
            const int nl  = nc0 + ct * 16 + l15;   // local j
            const float bm1 = bm1s[nl];
            const int jg  = j0 + nl;
            #pragma unroll
            for (int r = 0; r < 8; ++r) {
                const int ml = mr0 + rt * 16 + r + 8 * half;  // C/D: M=r+8*half
                float v = fmaf(acc[rt][ct][r], bm1, t1s[ml]);
                v = v > 0.f ? v : 0.f;
                if (i0 + ml == jg) v = 0.f;
                lsum += v;
            }
        }
    }
    red[t] = lsum;
    __syncthreads();
    #pragma unroll
    for (int s = 128; s > 0; s >>= 1) {
        if (t < s) red[t] += red[t + s];
        __syncthreads();
    }
    if (t == 0) partials[blockIdx.y * gridDim.x + blockIdx.x] = red[0];
}

// =====================================================================
// Kernel 3: deterministic final reduction of block partials.
// =====================================================================
__global__ __launch_bounds__(256)
void icc_finalize(const float* __restrict__ partials,
                  float* __restrict__ out, int np) {
    __shared__ float red[256];
    float s = 0.f;
    for (int k = threadIdx.x; k < np; k += 256) s += partials[k];
    red[threadIdx.x] = s;
    __syncthreads();
    #pragma unroll
    for (int st = 128; st > 0; st >>= 1) {
        if (threadIdx.x < st) red[threadIdx.x] += red[threadIdx.x + st];
        __syncthreads();
    }
    if (threadIdx.x == 0)
        out[0] = red[0] * (1.0f / ((float)NCLS * (float)(NCLS - 1)));
}

// =====================================================================
extern "C" void kernel_launch(void* const* d_in, const int* in_sizes, int n_in,
                              void* d_out, int out_size, void* d_ws, size_t ws_size,
                              hipStream_t stream) {
    const float* boundaries = (const float*)d_in[0];   // [4096]
    const float* prototypes = (const float*)d_in[1];   // [4096, 2048]
    const float* weights    = (const float*)d_in[2];   // [4096, 2048]
    float* out = (float*)d_out;

    const size_t NF = (size_t)NCLS * FEAT;
    unsigned short* pB   = (unsigned short*)d_ws;      // bf16 normalized prototypes
    unsigned short* wB   = pB + NF;                    // bf16 weights
    float* term1         = (float*)(wB + NF);          // [4096]
    float* partials      = term1 + NCLS;               // [1024]

    icc_prep<<<NCLS, 256, 0, stream>>>(boundaries, prototypes, weights,
                                       pB, wB, term1);

    dim3 grid(NCLS / 128, NCLS / 128);                 // 32 x 32 = 1024 blocks
    icc_gemm<<<grid, 256, 0, stream>>>(pB, wB, term1, boundaries, partials);

    icc_finalize<<<1, 256, 0, stream>>>(partials, out, 1024);
}
